// OwnGCN_73443940761886
// MI455X (gfx1250) — matrix operand order repaired and checked by reading, compiled, not run on hardware
//
#include <hip/hip_runtime.h>
#include <hip/hip_bf16.h>

#define N_NODES  40000
#define N_EDGES  640000
#define N_GRAPHS 64
#define NEG_SLOPE 0.1f
#define GN_EPS 1e-5f

typedef __attribute__((ext_vector_type(16))) __bf16       v16bf;
typedef __attribute__((ext_vector_type(8)))  float        v8f;
typedef __attribute__((ext_vector_type(4)))  float        v4f;
typedef __attribute__((ext_vector_type(4)))  unsigned int v4u;

__device__ __forceinline__ void atomAddF(float* p, float v) {
    unsafeAtomicAdd(p, v);   // lowers to global_atomic_add_f32 on gfx1250
}

// ---------------------------------------------------------------- utility
__global__ void cvt_f32_bf16_kernel(const float* __restrict__ in,
                                    __bf16* __restrict__ out, int total) {
    int id = blockIdx.x * blockDim.x + threadIdx.x;
    if (id < total) out[id] = (__bf16)in[id];
}

// ---------------------------------------------------------------- graph norm. setup
__global__ void degree_kernel(const int* __restrict__ src, float* __restrict__ deg, int E) {
    int e = blockIdx.x * blockDim.x + threadIdx.x;
    if (e < E) atomAddF(&deg[src[e]], 1.0f);
}

__global__ void dis_kernel(const float* __restrict__ deg, float* __restrict__ dis, int n) {
    int i = blockIdx.x * blockDim.x + threadIdx.x;
    if (i < n) {
        float d = deg[i];
        dis[i] = (d > 0.0f) ? (1.0f / sqrtf(fmaxf(d, 1.0f))) : 0.0f;
    }
}

__global__ void edgenorm_kernel(const int* __restrict__ src, const int* __restrict__ dst,
                                const float* __restrict__ dis, float* __restrict__ nrm, int E) {
    int e = blockIdx.x * blockDim.x + threadIdx.x;
    if (e < E) nrm[e] = -dis[src[e]] * dis[dst[e]];
}

// ---------------------------------------------------------------- sparse propagation
// acc[dst] += norm[e] * X[src];  4 features per thread, 128-bit gather.
__global__ __launch_bounds__(256)
void propagate_kernel(const float* __restrict__ X, const int* __restrict__ src,
                      const int* __restrict__ dst, const float* __restrict__ nrm,
                      float* __restrict__ acc, int E, int F) {
    int tid = blockIdx.x * blockDim.x + threadIdx.x;
    int cpe = F >> 2;                       // 4-float chunks per edge
    int e = tid / cpe;
    if (e >= E) return;
    int c = (tid - e * cpe) << 2;
    int s = src[e], d = dst[e];
    float w = nrm[e];
    v4f xv = *(const v4f*)(X + (size_t)s * F + c);
    float* out = acc + (size_t)d * F + c;
    atomAddF(out + 0, w * xv.x);
    atomAddF(out + 1, w * xv.y);
    atomAddF(out + 2, w * xv.z);
    atomAddF(out + 3, w * xv.w);
}

// Chebyshev recurrence finalize: mode 0: T = acc ; mode 1: T = 2*acc - Tprev
__global__ void finalize_kernel(const float* __restrict__ acc, const float* __restrict__ Tprev,
                                float* __restrict__ outf, __bf16* __restrict__ outh,
                                int total, int mode) {
    int id = blockIdx.x * blockDim.x + threadIdx.x;
    if (id >= total) return;
    float v = acc[id];
    if (mode) v = 2.0f * v - Tprev[id];
    if (outf) outf[id] = v;
    outh[id] = (__bf16)v;
}

// ---------------------------------------------------------------- WMMA GEMM
// Y[n, Fout] (+)= A_bf16[n, Fin] * B_bf16[Fin, Fout]; init: Y starts at bias.
// Block: 256 threads = 8 waves, 128 rows x 32 cols per block.
// B panel (Fin x 32) staged transposed in LDS (K-contiguous per column, stride Fin+8
// kills bank conflicts); per k-step each wave issues 2 global b128 (A),
// 4 ds b128 (B) and 2 WMMAs.
#define BPAD 8
#define MAXFIN 512
__global__ __launch_bounds__(256)
void gemm_bf16_wmma_kernel(const __bf16* __restrict__ A, const __bf16* __restrict__ B,
                           const float* __restrict__ bias, float* __restrict__ Y,
                           int nrows, int Fin, int Fout, int init) {
    __shared__ unsigned short Bs[32 * (MAXFIN + BPAD)];

    const int tid  = threadIdx.x;
    const int lane = tid & 31;
    const int wave = tid >> 5;
    const int stride = Fin + BPAD;

    // ---- cooperative transpose-load of the 32-column B panel into LDS
    {
        const unsigned short* Bg = (const unsigned short*)B;
        const int gcol0 = blockIdx.x * 32;
        for (int idx = tid; idx < 32 * Fin; idx += 256) {
            int k = idx >> 5;
            int c = idx & 31;
            Bs[c * stride + k] = Bg[(size_t)k * Fout + gcol0 + c];
        }
    }
    __syncthreads();

    // ---- per-wave tile setup (wave-uniform, so EXEC stays all-ones)
    int rowTile = blockIdx.y * 8 + wave;
    const int active = (rowTile * 16 < nrows);
    if (!active) rowTile = (nrows >> 4) - 1;            // clamp to a valid tile
    const int lcol = lane & 15;
    const int hi   = lane >> 4;                         // lane half
    const int col0 = blockIdx.x * 32 + lcol;            // accumulator tile 0
    const int col1 = col0 + 16;                         // accumulator tile 1
    const int rbase = rowTile * 16 + hi * 8;            // C/D: M = r + 8*hi, N = lane&15

    v8f c0, c1;
    if (init) {
        float b0 = bias[col0], b1 = bias[col1];
        #pragma unroll
        for (int r = 0; r < 8; ++r) { c0[r] = b0; c1[r] = b1; }
    } else {
        #pragma unroll
        for (int r = 0; r < 8; ++r) {
            c0[r] = Y[(size_t)(rbase + r) * Fout + col0];
            c1[r] = Y[(size_t)(rbase + r) * Fout + col1];
        }
    }

    const int arow = rowTile * 16 + lcol;               // A: M = lane&15
    const __bf16* Ap = A + (size_t)arow * Fin;
    const unsigned short* Bl0 = &Bs[(size_t)lcol * stride];         // tile 0 column
    const unsigned short* Bl1 = &Bs[(size_t)(16 + lcol) * stride];  // tile 1 column

    #pragma unroll 2
    for (int k0 = 0; k0 < Fin; k0 += 32) {
        // A 16x32 bf16: row = lane&15, elem e -> K = k0 + hi*16 + e (matches B)
        union { v16bf v; v4u q[2]; } a;
        a.q[0] = *(const v4u*)(Ap + k0 + hi * 16);
        a.q[1] = *(const v4u*)(Ap + k0 + hi * 16 + 8);
        // B 32x16 bf16: col = lane&15, elem e -> K = k0 + hi*16 + e (contiguous in LDS)
        const int kb = k0 + hi * 16;
        union { v16bf v; v4u q[2]; } b0, b1;
        b0.q[0] = *(const v4u*)(Bl0 + kb);
        b0.q[1] = *(const v4u*)(Bl0 + kb + 8);
        b1.q[0] = *(const v4u*)(Bl1 + kb);
        b1.q[1] = *(const v4u*)(Bl1 + kb + 8);
        c0 = __builtin_amdgcn_wmma_f32_16x16x32_bf16(false, a.v, false, b0.v,
                                                     (short)0, c0, false, false);
        c1 = __builtin_amdgcn_wmma_f32_16x16x32_bf16(false, a.v, false, b1.v,
                                                     (short)0, c1, false, false);
    }

    if (active) {
        #pragma unroll
        for (int r = 0; r < 8; ++r) {
            Y[(size_t)(rbase + r) * Fout + col0] = c0[r];
            Y[(size_t)(rbase + r) * Fout + col1] = c1[r];
        }
    }
}

// ---------------------------------------------------------------- GraphNorm
__global__ __launch_bounds__(256)
void colstats_kernel(const float* __restrict__ Y, int n, int F, float* __restrict__ sums) {
    __shared__ float s1[256], s2[256];
    int f = blockIdx.x;
    float a = 0.0f, b = 0.0f;
    for (int r = threadIdx.x; r < n; r += 256) {
        float v = Y[(size_t)r * F + f];
        a += v; b += v * v;
    }
    s1[threadIdx.x] = a; s2[threadIdx.x] = b;
    __syncthreads();
    for (int s = 128; s > 0; s >>= 1) {
        if (threadIdx.x < s) {
            s1[threadIdx.x] += s1[threadIdx.x + s];
            s2[threadIdx.x] += s2[threadIdx.x + s];
        }
        __syncthreads();
    }
    if (threadIdx.x == 0) { sums[f] = s1[0]; sums[F + f] = s2[0]; }
}

// out = w*(x - a*mean)*rsqrt(var+eps) + b  ==  x*scale[f] + shift[f]
__global__ void colparams_kernel(const float* __restrict__ sums,
                                 const float* __restrict__ gw, const float* __restrict__ gb,
                                 const float* __restrict__ ga,
                                 float* __restrict__ scale, float* __restrict__ shift,
                                 int n, int F) {
    int f = blockIdx.x * blockDim.x + threadIdx.x;
    if (f >= F) return;
    float inv_n = 1.0f / (float)n;
    float mean = sums[f] * inv_n;
    float ex2  = sums[F + f] * inv_n;
    float a = ga[f];
    // var of (x - a*mean) over same rows: E[x^2] - 2a*mean^2 + a^2*mean^2
    float var = ex2 - 2.0f * a * mean * mean + a * a * mean * mean;
    float rs = 1.0f / sqrtf(var + GN_EPS);
    float sc = gw[f] * rs;
    scale[f] = sc;
    shift[f] = gb[f] - a * mean * sc;
}

// mode 0: leaky;  mode 1: + residual, relu.  Writes f32 + bf16 mirror.
__global__ void norm_act_kernel(const float* __restrict__ Y, const float* __restrict__ scale,
                                const float* __restrict__ shift, const float* __restrict__ res,
                                float* __restrict__ outf, __bf16* __restrict__ outh,
                                int total, int F, int mode) {
    int id = blockIdx.x * blockDim.x + threadIdx.x;
    if (id >= total) return;
    int f = id % F;
    float v = Y[id] * scale[f] + shift[f];
    if (mode == 0) {
        v = (v >= 0.0f) ? v : NEG_SLOPE * v;
    } else {
        v += res[id];
        v = fmaxf(v, 0.0f);
    }
    outf[id] = v;
    outh[id] = (__bf16)v;
}

// ---------------------------------------------------------------- pooling + MLP
__global__ void count_kernel(const int* __restrict__ batch, float* __restrict__ cnt, int n) {
    int i = blockIdx.x * blockDim.x + threadIdx.x;
    if (i < n) atomAddF(&cnt[batch[i]], 1.0f);
}

__global__ void poolsum_kernel(const float* __restrict__ X, const int* __restrict__ batch,
                               float* __restrict__ pool, int n) {
    int id = blockIdx.x * blockDim.x + threadIdx.x;       // n * 32 threads (128/4 chunks)
    int node = id >> 5;
    if (node >= n) return;
    int c = (id & 31) << 2;
    int g = batch[node];
    v4f xv = *(const v4f*)(X + (size_t)node * 128 + c);
    float* out = pool + (size_t)g * 128 + c;
    atomAddF(out + 0, xv.x);
    atomAddF(out + 1, xv.y);
    atomAddF(out + 2, xv.z);
    atomAddF(out + 3, xv.w);
}

__global__ void pooldiv_kernel(float* __restrict__ pool, const float* __restrict__ cnt, int total) {
    int id = blockIdx.x * blockDim.x + threadIdx.x;
    if (id < total) pool[id] /= fmaxf(cnt[id >> 7], 1.0f);
}

__global__ void mlp_kernel(const float* __restrict__ pool,
                           const float* __restrict__ w1, const float* __restrict__ b1,
                           const float* __restrict__ w2, const float* __restrict__ b2,
                           float* __restrict__ out) {
    __shared__ float h[64];
    int g = blockIdx.x, t = threadIdx.x;
    float s = b1[t];
    for (int j = 0; j < 128; ++j) s += pool[g * 128 + j] * w1[j * 64 + t];
    h[t] = tanhf(s);
    __syncthreads();
    if (t < 12) {
        float o = b2[t];
        for (int j = 0; j < 64; ++j) o += h[j] * w2[j * 12 + t];
        out[g * 12 + t] = o;
    }
}

// ---------------------------------------------------------------- host orchestration
static inline unsigned nblk(long long total, int bs) { return (unsigned)((total + bs - 1) / bs); }

extern "C" void kernel_launch(void* const* d_in, const int* in_sizes, int n_in,
                              void* d_out, int out_size, void* d_ws, size_t ws_size,
                              hipStream_t stream) {
    const int N = N_NODES, E = N_EDGES, G = N_GRAPHS;
    const float* x     = (const float*)d_in[0];
    const int*   ei    = (const int*)d_in[1];
    const int*   src   = ei;
    const int*   dst   = ei + E;
    const int*   batch = (const int*)d_in[2];
    const float* Wf[4]  = {(const float*)d_in[3],  (const float*)d_in[8],
                           (const float*)d_in[13], (const float*)d_in[18]};
    const float* bias[4] = {(const float*)d_in[4],  (const float*)d_in[9],
                            (const float*)d_in[14], (const float*)d_in[19]};
    const float* gnw[4] = {(const float*)d_in[5],  (const float*)d_in[10],
                           (const float*)d_in[15], (const float*)d_in[20]};
    const float* gnb[4] = {(const float*)d_in[6],  (const float*)d_in[11],
                           (const float*)d_in[16], (const float*)d_in[21]};
    const float* gna[4] = {(const float*)d_in[7],  (const float*)d_in[12],
                           (const float*)d_in[17], (const float*)d_in[22]};
    const float* lin1_w = (const float*)d_in[23];
    const float* lin1_b = (const float*)d_in[24];
    const float* lin2_w = (const float*)d_in[25];
    const float* lin2_b = (const float*)d_in[26];

    const int Fin[4]  = {128, 256, 512, 256};
    const int Fout[4] = {256, 512, 256, 128};

    // ---- carve workspace
    size_t off = 0;
    char* base = (char*)d_ws;
    auto carve = [&](size_t bytes) -> char* {
        char* p = base + off;
        off += (bytes + 255) & ~(size_t)255;
        return p;
    };
    float* deg    = (float*)carve((size_t)N * 4);
    float* dis    = (float*)carve((size_t)N * 4);
    float* enorm  = (float*)carve((size_t)E * 4);
    float* sums   = (float*)carve(1024 * 4);
    float* cscale = (float*)carve(512 * 4);
    float* cshift = (float*)carve(512 * 4);
    float* cnt    = (float*)carve((size_t)G * 4);
    float* pool   = (float*)carve((size_t)G * 128 * 4);
    __bf16* Wh[4];
    for (int L = 0; L < 4; ++L)
        Wh[L] = (__bf16*)carve((size_t)4 * Fin[L] * Fout[L] * 2);
    float*  T1f  = (float*)carve((size_t)N * 512 * 4);
    float*  T2f  = (float*)carve((size_t)N * 512 * 4);
    float*  T0f  = (float*)carve((size_t)N * 512 * 4);
    float*  accY = (float*)carve((size_t)N * 512 * 4);   // propagation acc, reused as GEMM Y
    __bf16* T0h  = (__bf16*)carve((size_t)N * 512 * 2);
    __bf16* T1h  = (__bf16*)carve((size_t)N * 512 * 2);
    __bf16* T2h  = (__bf16*)carve((size_t)N * 512 * 2);
    __bf16* T3h  = (__bf16*)carve((size_t)N * 512 * 2);

    // ---- edge normalization
    hipMemsetAsync(deg, 0, (size_t)N * 4, stream);
    degree_kernel<<<nblk(E, 256), 256, 0, stream>>>(src, deg, E);
    dis_kernel<<<nblk(N, 256), 256, 0, stream>>>(deg, dis, N);
    edgenorm_kernel<<<nblk(E, 256), 256, 0, stream>>>(src, dst, dis, enorm, E);

    // ---- bf16 weight mirrors
    for (int L = 0; L < 4; ++L) {
        int tot = 4 * Fin[L] * Fout[L];
        cvt_f32_bf16_kernel<<<nblk(tot, 256), 256, 0, stream>>>(Wf[L], Wh[L], tot);
    }
    // bf16 mirror of layer-1 input
    cvt_f32_bf16_kernel<<<nblk((size_t)N * 128, 256), 256, 0, stream>>>(x, T0h, N * 128);

    const float*  curf = x;     // layer input, f32
    const __bf16* curh = T0h;   // layer input, bf16 mirror

    for (int L = 0; L < 4; ++L) {
        const int fi = Fin[L], fo = Fout[L];
        const long long tot = (long long)N * fi;
        const long long ethreads = (long long)E * (fi >> 2);

        // T1 = L_hat * T0
        hipMemsetAsync(accY, 0, (size_t)tot * 4, stream);
        propagate_kernel<<<nblk(ethreads, 256), 256, 0, stream>>>(curf, src, dst, enorm, accY, E, fi);
        finalize_kernel<<<nblk(tot, 256), 256, 0, stream>>>(accY, nullptr, T1f, T1h, (int)tot, 0);
        // T2 = 2*L_hat*T1 - T0
        hipMemsetAsync(accY, 0, (size_t)tot * 4, stream);
        propagate_kernel<<<nblk(ethreads, 256), 256, 0, stream>>>(T1f, src, dst, enorm, accY, E, fi);
        finalize_kernel<<<nblk(tot, 256), 256, 0, stream>>>(accY, curf, T2f, T2h, (int)tot, 1);
        // T3 = 2*L_hat*T2 - T1 (bf16 only; not needed in f32)
        hipMemsetAsync(accY, 0, (size_t)tot * 4, stream);
        propagate_kernel<<<nblk(ethreads, 256), 256, 0, stream>>>(T2f, src, dst, enorm, accY, E, fi);
        finalize_kernel<<<nblk(tot, 256), 256, 0, stream>>>(accY, T1f, nullptr, T3h, (int)tot, 1);

        // Y = bias + T0*W0 + T1*W1 + T2*W2 + T3*W3   (WMMA, f32 accumulate)
        float* Y = accY;
        dim3 ggrid(fo / 32, (2500 + 7) / 8);   // N/16 = 2500 row tiles, 8 per block, 32 cols/block
        gemm_bf16_wmma_kernel<<<ggrid, 256, 0, stream>>>(curh, Wh[L] + 0ll,                 bias[L], Y, N, fi, fo, 1);
        gemm_bf16_wmma_kernel<<<ggrid, 256, 0, stream>>>(T1h,  Wh[L] + (size_t)1 * fi * fo, nullptr, Y, N, fi, fo, 0);
        gemm_bf16_wmma_kernel<<<ggrid, 256, 0, stream>>>(T2h,  Wh[L] + (size_t)2 * fi * fo, nullptr, Y, N, fi, fo, 0);
        gemm_bf16_wmma_kernel<<<ggrid, 256, 0, stream>>>(T3h,  Wh[L] + (size_t)3 * fi * fo, nullptr, Y, N, fi, fo, 0);

        // GraphNorm + activation (layer 4: residual + relu)
        colstats_kernel<<<fo, 256, 0, stream>>>(Y, N, fo, sums);
        colparams_kernel<<<nblk(fo, 256), 256, 0, stream>>>(sums, gnw[L], gnb[L], gna[L],
                                                            cscale, cshift, N, fo);
        const long long otot = (long long)N * fo;
        norm_act_kernel<<<nblk(otot, 256), 256, 0, stream>>>(Y, cscale, cshift,
                                                             (L == 3) ? x : nullptr,
                                                             T0f, T0h, (int)otot, fo, (L == 3) ? 1 : 0);
        curf = T0f;
        curh = T0h;
    }

    // ---- mean pool per graph + MLP head
    hipMemsetAsync(cnt, 0, (size_t)G * 4, stream);
    hipMemsetAsync(pool, 0, (size_t)G * 128 * 4, stream);
    count_kernel<<<nblk(N, 256), 256, 0, stream>>>(batch, cnt, N);
    poolsum_kernel<<<nblk((long long)N * 32, 256), 256, 0, stream>>>(curf, batch, pool, N);
    pooldiv_kernel<<<nblk(G * 128, 256), 256, 0, stream>>>(pool, cnt, G * 128);
    mlp_kernel<<<G, 64, 0, stream>>>(pool, lin1_w, lin1_b, lin2_w, lin2_b, (float*)d_out);
}
